// GCN_13082470384334
// MI455X (gfx1250) — matrix operand order repaired and checked by reading, not compiled, not run on hardware
//
#include <hip/hip_runtime.h>
#include <hip/hip_bf16.h>

typedef __attribute__((ext_vector_type(16))) __bf16 v16bf;
typedef __attribute__((ext_vector_type(2)))  __bf16 v2bf;
typedef __attribute__((ext_vector_type(8)))  float  v8f;

#define NNODES 50000
#define NEDGES 800000
#define NGRAPH 100

#if defined(__has_builtin)
#if __has_builtin(__builtin_amdgcn_cvt_pk_bf16_f32)
#define GCN_HAVE_CVT_PK_BF16 1
#endif
#endif

static __device__ __forceinline__ unsigned pack_bf16(float a, float b) {
  // packed bf16 pair word: low = a, high = b (single v_cvt_pk_bf16_f32)
#ifdef GCN_HAVE_CVT_PK_BF16
  v2bf v = __builtin_amdgcn_cvt_pk_bf16_f32(a, b);
  unsigned r;
  __builtin_memcpy(&r, &v, sizeof(r));
  return r;
#else
  __hip_bfloat162 h;
  h.x = __float2bfloat16(a);
  h.y = __float2bfloat16(b);
  unsigned r;
  __builtin_memcpy(&r, &h, sizeof(r));
  return r;
#endif
}

// ---------------------------------------------------------------- utilities
__global__ void gcn_zero(float* __restrict__ p, int n) {
  int i = blockIdx.x * blockDim.x + threadIdx.x;
  if (i < n) p[i] = 0.0f;
}

__global__ void gcn_degrees(const int* __restrict__ src, const int* __restrict__ dst,
                            float* __restrict__ deg_out, float* __restrict__ deg_in, int e) {
  int i = blockIdx.x * blockDim.x + threadIdx.x;
  if (i < e) {
    atomicAdd(&deg_out[src[i]], 1.0f);
    atomicAdd(&deg_in[dst[i]], 1.0f);
  }
}

__global__ void gcn_rnorm(float* __restrict__ p, int n) {
  int i = blockIdx.x * blockDim.x + threadIdx.x;
  if (i < n) p[i] = rsqrtf(fmaxf(p[i], 1.0f));
}

// One-shot: W[K x Nout] f32 -> Wt[Nout x K/2] packed bf16 pair words
// word (n, kp) = { W[2kp][n] (lo), W[2kp+1][n] (hi) }
__global__ void gcn_pack_w(const float* __restrict__ W, unsigned* __restrict__ Wt,
                           int Kp, int Nout) {
  int kp = blockIdx.x * blockDim.x + threadIdx.x;
  int n  = blockIdx.y;
  if (kp >= Kp) return;
  const float* p = W + (size_t)(2 * kp) * Nout + n;
  Wt[(size_t)n * Kp + kp] = pack_bf16(p[0], p[Nout]);
}

// ------------------------------------------------- WMMA bf16 GEMM with norm
// Out[N x Nout] = Ascaled[N x K] @ W[K x Nout]   (bf16 math, f32 accumulate)
// APACKED=0: A is f32, scaled by norm[row] during staging (layer 1, A = x).
// APACKED=1: A is pre-packed bf16 pair words [N x K/2], norm already folded.
// W always pre-packed transposed words Wt[Nout x K/2].
// Block: 256 threads = 8 waves; tile 64 rows x 64 cols; 2 WMMAs/wave/K-step.
// LDS ping-pong: one barrier per K-step.
template <bool APACKED>
__global__ __launch_bounds__(256) void gcn_gemm(
    const void* __restrict__ Ain, const unsigned* __restrict__ Wt,
    const float* __restrict__ norm, float* __restrict__ Out,
    int N, int K, int Nout)
{
  __shared__ unsigned ldsA[2 * 64 * 16];   // 64 rows x 16 k-pairs, double-buffered
  __shared__ unsigned ldsB[2 * 64 * 20];   // 64 cols x 16 k-pairs (pad 20), double-buffered

  const int t    = threadIdx.x;
  const int lane = t & 31;
  const int wave = t >> 5;
  const int rt   = wave & 3;           // row tile (0..3)
  const int ctb  = (wave >> 2) * 2;    // first of two col tiles (0 or 2)
  const int rowBase = blockIdx.x * 64;
  const int colBase = blockIdx.y * 64;
  const int l16 = lane & 15;
  const int lhi = lane >> 4;
  const int Kp  = K >> 1;

  // ---- loop-invariant staging state (thread t -> row t>>2, word base (t&3)*4) ----
  const int sRow = t >> 2;             // 0..63
  const int sW4  = (t & 3) * 4;        // 0,4,8,12

  const int  grow = rowBase + sRow;
  const bool aok  = grow < N;
  const int  crow = aok ? grow : 0;    // clamped row (stays in-bounds)
  // f32 path: scale 0 for OOB rows -> loaded garbage * 0 == 0, branch-free
  const float     as    = (!APACKED && aok) ? norm[grow] : 0.0f;
  const float*    apf   = (const float*)Ain + (size_t)crow * K + sW4 * 2;
  const unsigned* apk   = (const unsigned*)Ain + (size_t)crow * Kp + sW4;
  const unsigned* bwp   = Wt + (size_t)(colBase + sRow) * Kp + sW4;

  v8f c0 = {0.f, 0.f, 0.f, 0.f, 0.f, 0.f, 0.f, 0.f};
  v8f c1 = {0.f, 0.f, 0.f, 0.f, 0.f, 0.f, 0.f, 0.f};

  int sel = 0;
  for (int k0 = 0; k0 < K; k0 += 32, sel ^= 1) {
    unsigned* lA = ldsA + sel * (64 * 16);
    unsigned* lB = ldsB + sel * (64 * 20);
    const int k0p = k0 >> 1;

    // Stage A tile 64x32 -> one ds_store_b128 per thread
    uint4 aw;
    if (APACKED) {
      aw = make_uint4(0u, 0u, 0u, 0u);
      if (aok) aw = *(const uint4*)(apk + k0p);
    } else {
      const float4 lo = *(const float4*)(apf + k0);
      const float4 hi = *(const float4*)(apf + k0 + 4);
      aw.x = pack_bf16(lo.x * as, lo.y * as);
      aw.y = pack_bf16(lo.z * as, lo.w * as);
      aw.z = pack_bf16(hi.x * as, hi.y * as);
      aw.w = pack_bf16(hi.z * as, hi.w * as);
    }
    *(uint4*)&lA[sRow * 16 + sW4] = aw;

    // Stage W tile 64n x 32k (pre-packed): b128 load -> b128 store
    *(uint4*)&lB[sRow * 20 + sW4] = *(const uint4*)(bwp + k0p);

    __syncthreads();   // ping-pong: single barrier per step

    // A fragment (16x32): lane l16 = M; VGPR v -> K-pair lhi*8 + v
    union { v16bf v; unsigned u[8]; } af, bf0, bf1;
    const int am = rt * 16 + l16;
#pragma unroll
    for (int v = 0; v < 8; ++v)
      af.u[v] = lA[am * 16 + lhi * 8 + v];
    // B fragments (32x16): lane l16 = N; VGPR v -> K-pair lhi*8 + v (b128-able)
    const int bna = (ctb + 0) * 16 + l16;
    const int bnb = (ctb + 1) * 16 + l16;
#pragma unroll
    for (int v = 0; v < 8; ++v) {
      bf0.u[v] = lB[bna * 20 + lhi * 8 + v];
      bf1.u[v] = lB[bnb * 20 + lhi * 8 + v];
    }

    c0 = __builtin_amdgcn_wmma_f32_16x16x32_bf16(false, af.v, false, bf0.v,
                                                 (short)0, c0, false, false);
    c1 = __builtin_amdgcn_wmma_f32_16x16x32_bf16(false, af.v, false, bf1.v,
                                                 (short)0, c1, false, false);
  }

  // C/D layout: lane l16 = N; VGPR r -> M = r + lhi*8
  const int col0 = colBase + (ctb + 0) * 16 + l16;
  const int col1 = colBase + (ctb + 1) * 16 + l16;
  const int rowTop = rowBase + rt * 16 + lhi * 8;
  if (rowBase + 64 <= N) {               // fast path: whole tile in range
    float* p = Out + (size_t)rowTop * Nout;
#pragma unroll
    for (int r = 0; r < 8; ++r) {
      p[col0] = c0[r];
      p[col1] = c1[r];
      p += Nout;
    }
  } else {
#pragma unroll
    for (int r = 0; r < 8; ++r) {
      int row = rowTop + r;
      if (row < N) {
        Out[(size_t)row * Nout + col0] = c0[r];
        Out[(size_t)row * Nout + col1] = c1[r];
      }
    }
  }
}

// ------------------------------------------------------------- SpMM scatter
// out[dst[e]] += h[src[e]]  (edge-parallel, float4 per thread, atomics hit L2)
__global__ void gcn_spmm(const float* __restrict__ h, const int* __restrict__ src,
                         const int* __restrict__ dst, float* __restrict__ out,
                         int e, int F, int pshift) {
  long long g = (long long)blockIdx.x * blockDim.x + threadIdx.x;
  int ed = (int)(g >> pshift);
  int cc = (int)(g & ((1 << pshift) - 1));
  if (ed >= e) return;
  const float4 v = ((const float4*)(h + (size_t)src[ed] * F))[cc];
  float* o = out + (size_t)dst[ed] * F + cc * 4;
  atomicAdd(o + 0, v.x);
  atomicAdd(o + 1, v.y);
  atomicAdd(o + 2, v.z);
  atomicAdd(o + 3, v.w);
}

// h = relu(m * norm_in[row] + b[col]); emit packed bf16 words scaled by
// norm_out[row] (folds next layer's A-side normalization + conversion).
__global__ void gcn_finalize_pack(const float* __restrict__ m,
                                  const float* __restrict__ norm_in,
                                  const float* __restrict__ norm_out,
                                  const float* __restrict__ b,
                                  unsigned* __restrict__ outw,
                                  int N, int fshift) {
  const int ws = fshift - 1;                       // words per row = F/2
  int i = blockIdx.x * blockDim.x + threadIdx.x;
  if (i >= (N << ws)) return;
  int row = i >> ws, c = i & ((1 << ws) - 1);
  const float2 mv = ((const float2*)(m + ((size_t)row << fshift)))[c];
  float ni = norm_in[row], no = norm_out[row];
  float v0 = fmaxf(mv.x * ni + b[2 * c + 0], 0.0f) * no;
  float v1 = fmaxf(mv.y * ni + b[2 * c + 1], 0.0f) * no;
  outw[i] = pack_bf16(v0, v1);
}

// relu(m * norm_in[row] + b[col]) in place (last layer, feeds f32 pooling)
__global__ void gcn_finalize(float* __restrict__ m, const float* __restrict__ norm_in,
                             const float* __restrict__ b, int N, int fshift) {
  int i = blockIdx.x * blockDim.x + threadIdx.x;
  if (i >= (N << fshift)) return;
  int row = i >> fshift, col = i & ((1 << fshift) - 1);
  m[i] = fmaxf(m[i] * norm_in[row] + b[col], 0.0f);
}

// ----------------------------------------------------------------- pooling
__global__ void gcn_pool(const float* __restrict__ h, const int* __restrict__ gid,
                         float* __restrict__ pool, float* __restrict__ cnt, int N) {
  int i = blockIdx.x * blockDim.x + threadIdx.x;
  if (i >= N * 64) return;
  int n = i >> 6, f = i & 63;
  int g = gid[n];
  atomicAdd(&pool[g * 64 + f], h[i]);
  if (f == 0) atomicAdd(&cnt[g], 1.0f);
}

__global__ void gcn_fc(const float* __restrict__ pool, const float* __restrict__ cnt,
                       const float* __restrict__ Wfc, const float* __restrict__ bfc,
                       float* __restrict__ out) {
  int i = threadIdx.x;
  if (i >= NGRAPH * 2) return;
  int g = i >> 1, c = i & 1;
  float inv = 1.0f / fmaxf(cnt[g], 1.0f);
  float acc = bfc[c];
#pragma unroll
  for (int f = 0; f < 64; ++f) acc += pool[g * 64 + f] * inv * Wfc[f * 2 + c];
  out[i] = acc;
}

// ----------------------------------------------------------------- launcher
extern "C" void kernel_launch(void* const* d_in, const int* in_sizes, int n_in,
                              void* d_out, int out_size, void* d_ws, size_t ws_size,
                              hipStream_t stream) {
  const float* x   = (const float*)d_in[0];
  const int*   src = (const int*)d_in[1];
  const int*   dst = (const int*)d_in[2];
  const int*   gid = (const int*)d_in[3];
  const float* W1  = (const float*)d_in[4];
  const float* b1  = (const float*)d_in[5];
  const float* W2  = (const float*)d_in[6];
  const float* b2  = (const float*)d_in[7];
  const float* W3  = (const float*)d_in[8];
  const float* b3  = (const float*)d_in[9];
  const float* Wfc = (const float*)d_in[10];
  const float* bfc = (const float*)d_in[11];
  float* out = (float*)d_out;

  const int N = NNODES, E = NEDGES;

  // carve workspace (256B aligned slabs)
  size_t off = 0;
  auto carve = [&](size_t bytes) {
    off = (off + 255) & ~(size_t)255;
    void* p = (char*)d_ws + off;
    off += bytes;
    return p;
  };
  float*    norm_out = (float*)carve((size_t)N * 4);
  float*    norm_in  = (float*)carve((size_t)N * 4);
  unsigned* W1t      = (unsigned*)carve((size_t)256 * 384 * 4);
  unsigned* W2t      = (unsigned*)carve((size_t)128 * 128 * 4);
  unsigned* W3t      = (unsigned*)carve((size_t)64 * 64 * 4);
  float*    bufP     = (float*)carve((size_t)N * 256 * 4);    // GEMM out (pre-agg)
  float*    bufM     = (float*)carve((size_t)N * 256 * 4);    // SpMM accumulator
  unsigned* Hbf      = (unsigned*)carve((size_t)N * 128 * 4); // packed activations
  float*    poolb    = (float*)carve((size_t)NGRAPH * 64 * 4);
  float*    cnt      = (float*)carve((size_t)NGRAPH * 4);
  (void)ws_size; (void)in_sizes; (void)n_in; (void)out_size;

  const int T = 256;
  auto blocks = [](long long n, int t) { return (unsigned)((n + t - 1) / t); };

  // degrees -> symmetric norms
  gcn_zero<<<blocks(N, T), T, 0, stream>>>(norm_out, N);
  gcn_zero<<<blocks(N, T), T, 0, stream>>>(norm_in, N);
  gcn_degrees<<<blocks(E, T), T, 0, stream>>>(src, dst, norm_out, norm_in, E);
  gcn_rnorm<<<blocks(N, T), T, 0, stream>>>(norm_out, N);
  gcn_rnorm<<<blocks(N, T), T, 0, stream>>>(norm_in, N);

  // one-shot weight pack/transpose
  gcn_pack_w<<<dim3(blocks(384, T), 256), T, 0, stream>>>(W1, W1t, 384, 256);
  gcn_pack_w<<<dim3(blocks(128, T), 128), T, 0, stream>>>(W2, W2t, 128, 128);
  gcn_pack_w<<<dim3(blocks(64, T), 64),   T, 0, stream>>>(W3, W3t, 64, 64);

  const unsigned rowGrid = blocks(N, 64);

  // ---- layer 1: 768 -> 256 (A = x f32, norm folded at staging)
  gcn_gemm<false><<<dim3(rowGrid, 4), T, 0, stream>>>(x, W1t, norm_out, bufP, N, 768, 256);
  gcn_zero<<<blocks((long long)N * 256, T), T, 0, stream>>>(bufM, N * 256);
  gcn_spmm<<<blocks((long long)E * 64, T), T, 0, stream>>>(bufP, src, dst, bufM, E, 256, 6);
  gcn_finalize_pack<<<blocks((long long)N * 128, T), T, 0, stream>>>(bufM, norm_in, norm_out, b1, Hbf, N, 8);

  // ---- layer 2: 256 -> 128 (A = packed bf16)
  gcn_gemm<true><<<dim3(rowGrid, 2), T, 0, stream>>>(Hbf, W2t, nullptr, bufP, N, 256, 128);
  gcn_zero<<<blocks((long long)N * 128, T), T, 0, stream>>>(bufM, N * 128);
  gcn_spmm<<<blocks((long long)E * 32, T), T, 0, stream>>>(bufP, src, dst, bufM, E, 128, 5);
  gcn_finalize_pack<<<blocks((long long)N * 64, T), T, 0, stream>>>(bufM, norm_in, norm_out, b2, Hbf, N, 7);

  // ---- layer 3: 128 -> 64 (A = packed bf16)
  gcn_gemm<true><<<dim3(rowGrid, 1), T, 0, stream>>>(Hbf, W3t, nullptr, bufP, N, 128, 64);
  gcn_zero<<<blocks((long long)N * 64, T), T, 0, stream>>>(bufM, N * 64);
  gcn_spmm<<<blocks((long long)E * 16, T), T, 0, stream>>>(bufP, src, dst, bufM, E, 64, 4);
  gcn_finalize<<<blocks((long long)N * 64, T), T, 0, stream>>>(bufM, norm_in, b3, N, 6);

  // ---- mean pool + FC
  gcn_zero<<<blocks(NGRAPH * 64, T), T, 0, stream>>>(poolb, NGRAPH * 64);
  gcn_zero<<<1, T, 0, stream>>>(cnt, NGRAPH);
  gcn_pool<<<blocks((long long)N * 64, T), T, 0, stream>>>(bufM, gid, poolb, cnt, N);
  gcn_fc<<<1, 256, 0, stream>>>(poolb, cnt, Wfc, bfc, out);
}
